// LanguageContinuousThoughtMachine_79482664780483
// MI455X (gfx1250) — matrix operand order repaired and checked by reading, not compiled, not run on hardware
//
#include <hip/hip_runtime.h>

// ---------------------------------------------------------------------------
// LanguageContinuousThoughtMachine — CDNA5 (gfx1250) implementation.
// bf16 WMMA GEMMs (v_wmma_f32_16x16x32_bf16). 128x64 block tile, 4 waves,
// 8 WMMA / K-step / wave; B operand double-buffered through LDS with
// global_load_async_to_lds_b128 (ASYNCcnt), A software-pipelined in VGPRs.
// RoPE(K/V) hoisted out of the 8-step loop.
// ---------------------------------------------------------------------------

typedef unsigned int   u32;
typedef unsigned short u16;
typedef __attribute__((ext_vector_type(16))) __bf16 v16bf;
typedef __attribute__((ext_vector_type(8)))  float  v8f;
typedef __attribute__((ext_vector_type(4)))  u32    u32x4;

union V16U { v16bf v; u32x4 q[2]; };

// ---- model constants (from setup_inputs) ----
constexpr int CB   = 8;      // batch
constexpr int CL   = 4096;   // seq len
constexpr int CDIN = 1024;   // model dim
constexpr int CW   = 2048;   // neurons
constexpr int CDD  = 16;     // history depth D
constexpr int CQ   = 16;     // output queries
constexpr int CS   = 1024;   // sync pairs S
constexpr int CV   = 32000;  // vocab
constexpr int CHS  = 4096;   // syn hidden
constexpr int CNH  = 16;     // heads
constexpr int CDH  = 64;     // head dim
constexpr int CT   = 8;      // thought steps
constexpr int CBQ  = CB * CQ;          // 128
constexpr int CCAT = CDIN + CW;        // 3072

__device__ inline u16 f2bf(float f) {
  u32 u = __float_as_uint(f);
  u = (u + 0x7FFFu + ((u >> 16) & 1u)) >> 16;
  return (u16)u;
}
__device__ inline float bf2f(u16 h) { return __uint_as_float(((u32)h) << 16); }

__device__ inline float gelu_f(float x) {
  float x3 = x * x * x;
  return 0.5f * x * (1.0f + tanhf(0.7978845608f * (x + 0.044715f * x3)));
}

// CDNA5 async global->LDS copy (16B per lane), tracked by ASYNCcnt.
__device__ inline void async_b128(u32 lds_byte_off, const u16* src) {
  asm volatile("global_load_async_to_lds_b128 %0, %1, off"
               :: "v"(lds_byte_off), "v"(src) : "memory");
}
__device__ inline void wait_async_le2() {
  asm volatile("s_wait_asynccnt 2" ::: "memory");
}
__device__ inline void wait_async_le0() {
  asm volatile("s_wait_asynccnt 0" ::: "memory");
}

// ---------------------------------------------------------------------------
// GEMM: C[M,N](f32) = A[M,K](bf16, row-major) * BT[N,K](bf16, i.e. B^T) + bias
// Block tile: 128(M) x 64(N), 4 waves; wave w owns rows [w*32, w*32+32).
// Each wave: 2 (M) x 4 (N) accumulators -> 8 v_wmma_f32_16x16x32_bf16 / K-step.
// B tile (64 cols x 32 k = 4KB) staged in LDS, double buffered via async DMA;
// A fragments double-buffered in registers (loads issued before the WMMAs).
// ---------------------------------------------------------------------------
__global__ __launch_bounds__(128) void k_gemm_bf16(
    const u16* __restrict__ A, const u16* __restrict__ BT,
    const float* __restrict__ bias, float* __restrict__ C,
    int M, int N, int K, int addbias)
{
  __shared__ alignas(32) u16 ldsB[2][2048];   // 2 x 4KB: [col 0..63][32 k vals]
  const int tid  = threadIdx.x;
  const int wave = tid >> 5;
  const int lane = tid & 31;
  const int half = lane >> 4;     // 0: lanes 0-15, 1: lanes 16-31
  const int l    = lane & 15;

  const int ntn = N >> 6;                 // N / 64
  const int tm  = blockIdx.x / ntn;
  const int tn  = blockIdx.x % ntn;

  // --- async B staging: thread copies chunks tid and tid+128 (16B each) ---
  // chunk c: col = c>>2, sub = c&3 -> src BT[(tn*64+col)*K + k0 + sub*8]
  const int c0col = tid >> 2, c0sub = tid & 3;
  const int c1col = (tid + 128) >> 2, c1sub = (tid + 128) & 3;
  const u16* bsrc0 = BT + (size_t)(tn * 64 + c0col) * K + c0sub * 8;
  const u16* bsrc1 = BT + (size_t)(tn * 64 + c1col) * K + c1sub * 8;
  const u32 dst0[2] = {
    (u32)(unsigned long long)(void*)&ldsB[0][0] + (u32)tid * 16,
    (u32)(unsigned long long)(void*)&ldsB[1][0] + (u32)tid * 16 };
  const u32 dst1[2] = { dst0[0] + 2048, dst0[1] + 2048 };

  // --- A rows for this wave ---
  const u16* arow0 = A + (size_t)(tm * 128 + wave * 32 + l) * K;       // m-sub 0
  const u16* arow1 = arow0 + (size_t)16 * K;                           // m-sub 1

  // prologue: stage B tile for ki=0 into buffer 0; preload A for ki=0
  async_b128(dst0[0], bsrc0);
  async_b128(dst1[0], bsrc1);
  V16U a0c, a1c, a0n, a1n;
  {
    const int ka = half * 8;
    a0c.q[0] = *(const u32x4*)(arow0 + ka);
    a0c.q[1] = *(const u32x4*)(arow0 + ka + 16);
    a1c.q[0] = *(const u32x4*)(arow1 + ka);
    a1c.q[1] = *(const u32x4*)(arow1 + ka + 16);
  }

  v8f acc[2][4];
#pragma unroll
  for (int ms = 0; ms < 2; ++ms)
#pragma unroll
    for (int ns = 0; ns < 4; ++ns)
      acc[ms][ns] = (v8f){0.f, 0.f, 0.f, 0.f, 0.f, 0.f, 0.f, 0.f};

  const int nk = K >> 5;
  for (int ki = 0; ki < nk; ++ki) {
    const int cur = ki & 1, nxt = cur ^ 1;
    __syncthreads();                       // prev reads of `nxt` finished
    if (ki + 1 < nk) {
      const int kn = (ki + 1) << 5;
      async_b128(dst0[nxt], bsrc0 + kn);
      async_b128(dst1[nxt], bsrc1 + kn);
      wait_async_le2();                    // own copies for `cur` complete
      // A for ki+1: issue loads now, consumed after this step's WMMAs
      const int ka = kn + half * 8;
      a0n.q[0] = *(const u32x4*)(arow0 + ka);
      a0n.q[1] = *(const u32x4*)(arow0 + ka + 16);
      a1n.q[0] = *(const u32x4*)(arow1 + ka);
      a1n.q[1] = *(const u32x4*)(arow1 + ka + 16);
      __builtin_prefetch(arow0 + kn + 32, 0, 1);
    } else {
      wait_async_le0();
    }
    __syncthreads();                       // whole B tile `cur` visible

    // B fragments: col = ns*16 + l, k = half*8 + [0,8) and 16 + half*8 + [0,8)
    V16U b[4];
#pragma unroll
    for (int ns = 0; ns < 4; ++ns) {
      const u16* bp = &ldsB[cur][(ns * 16 + l) * 32 + half * 8];
      b[ns].q[0] = *(const u32x4*)(bp);
      b[ns].q[1] = *(const u32x4*)(bp + 16);
    }
#pragma unroll
    for (int ns = 0; ns < 4; ++ns) {
      acc[0][ns] = __builtin_amdgcn_wmma_f32_16x16x32_bf16(
          false, a0c.v, false, b[ns].v, (short)0, acc[0][ns], false, false);
      acc[1][ns] = __builtin_amdgcn_wmma_f32_16x16x32_bf16(
          false, a1c.v, false, b[ns].v, (short)0, acc[1][ns], false, false);
    }
    if (ki + 1 < nk) { a0c = a0n; a1c = a1n; }
  }

  // D layout: M = r + 8*half (+16*ms), N = l (+16*ns)
#pragma unroll
  for (int ms = 0; ms < 2; ++ms)
#pragma unroll
    for (int ns = 0; ns < 4; ++ns) {
      const int gn = tn * 64 + ns * 16 + l;
      const float bv = addbias ? bias[gn] : 0.0f;
      const int gm0 = tm * 128 + wave * 32 + ms * 16 + half * 8;
#pragma unroll
      for (int r = 0; r < 8; ++r)
        C[(size_t)(gm0 + r) * N + gn] = acc[ms][ns][r] + bv;
    }
}

// ---------------------------------------------------------------------------
// Small elementwise / setup kernels
// ---------------------------------------------------------------------------
__global__ void k_transpose_bf16(const float* __restrict__ in, u16* __restrict__ out,
                                 int K, int N) {
  size_t idx = (size_t)blockIdx.x * 256 + threadIdx.x;
  if (idx >= (size_t)K * N) return;
  int n = (int)(idx % N); int k = (int)(idx / N);
  out[(size_t)n * K + k] = f2bf(in[idx]);
}

__global__ void k_gather_embed(const int* __restrict__ x, const float* __restrict__ emb,
                               u16* __restrict__ out) {
  size_t idx = (size_t)blockIdx.x * 256 + threadIdx.x;
  if (idx >= (size_t)CB * CL * CDIN) return;
  size_t row = idx / CDIN; int d = (int)(idx % CDIN);
  int tok = x[row];
  out[idx] = f2bf(emb[(size_t)tok * CDIN + d]);
}

__global__ void k_cast_bf16(const float* __restrict__ in, u16* __restrict__ out, int n) {
  int idx = blockIdx.x * 256 + threadIdx.x;
  if (idx < n) out[idx] = f2bf(in[idx]);
}

__global__ void k_gelu_cast(const float* __restrict__ in, u16* __restrict__ out, int n) {
  int idx = blockIdx.x * 256 + threadIdx.x;
  if (idx < n) out[idx] = f2bf(gelu_f(in[idx]));
}

__global__ void k_concat_cast(const float* __restrict__ ap, const float* __restrict__ z,
                              u16* __restrict__ cat) {
  int idx = blockIdx.x * 256 + threadIdx.x;
  if (idx >= CBQ * CCAT) return;
  int r = idx / CCAT, c = idx % CCAT;
  float v = (c < CDIN) ? ap[r * CDIN + c] : z[r * CW + (c - CDIN)];
  cat[idx] = f2bf(v);
}

__global__ void k_init_hist(const float* __restrict__ init, float* __restrict__ hist) {
  size_t idx = (size_t)blockIdx.x * 256 + threadIdx.x;
  if (idx >= (size_t)CBQ * CW * CDD) return;
  hist[idx] = init[idx % ((size_t)CW * CDD)];
}

__global__ void k_init_post(const float* __restrict__ z0, float* __restrict__ ring0) {
  int idx = blockIdx.x * 256 + threadIdx.x;
  if (idx >= CBQ * CW) return;
  ring0[idx] = z0[idx % CW];
}

// RoPE for K, plus bf16 V, laid out (B,H,L,dh) for attention.
__global__ void k_rope_kv(const float* __restrict__ kvf, u16* __restrict__ kbf,
                          u16* __restrict__ vbf) {
  size_t idx = (size_t)blockIdx.x * 256 + threadIdx.x;
  if (idx >= (size_t)CB * CL * CNH * 32) return;
  int j = (int)(idx & 31);
  int h = (int)((idx >> 5) & (CNH - 1));
  int l = (int)((idx >> 9) & (CL - 1));
  int b = (int)(idx >> 21);
  const float* src = kvf + ((size_t)b * CL + l) * CDIN + h * CDH;
  float x1 = src[j], x2 = src[32 + j];
  float inv = __expf(-(float)j * 0.28782313662f);   // ln(10000)/32
  float ang = (float)l * inv;
  float c = __cosf(ang), si = __sinf(ang);
  size_t dst = (((size_t)b * CNH + h) * CL + l) * CDH;
  kbf[dst + j]      = f2bf(x1 * c - x2 * si);
  kbf[dst + 32 + j] = f2bf(x1 * si + x2 * c);
  vbf[dst + j]      = f2bf(x1);
  vbf[dst + 32 + j] = f2bf(x2);
}

// RoPE for q (pos = query index), (B,Q,Din) -> (B,H,Q,dh) bf16
__global__ void k_rope_q(const float* __restrict__ qf, u16* __restrict__ qbf) {
  int idx = blockIdx.x * 256 + threadIdx.x;
  if (idx >= CB * CQ * CNH * 32) return;
  int j = idx & 31;
  int h = (idx >> 5) & (CNH - 1);
  int q = (idx >> 9) & (CQ - 1);
  int b = idx >> 13;
  const float* src = qf + ((size_t)b * CQ + q) * CDIN + h * CDH;
  float x1 = src[j], x2 = src[32 + j];
  float inv = __expf(-(float)j * 0.28782313662f);
  float ang = (float)q * inv;
  float c = __cosf(ang), si = __sinf(ang);
  size_t dst = (((size_t)b * CNH + h) * CQ + q) * CDH;
  qbf[dst + j]      = f2bf(x1 * c - x2 * si);
  qbf[dst + 32 + j] = f2bf(x1 * si + x2 * c);
}

// Attention: one block per (b,h); 256 threads = 16 queries x 16 key stripes,
// online softmax per stripe, LDS merge. Causal mask: l <= L-Q+q.
__global__ __launch_bounds__(256) void k_attention(
    const u16* __restrict__ qbf, const u16* __restrict__ kbf,
    const u16* __restrict__ vbf, float* __restrict__ out) {
  const int bh = blockIdx.x;
  const int b = bh / CNH, h = bh % CNH;
  const int tid = threadIdx.x;
  const int q = tid & 15, stripe = tid >> 4;

  __shared__ float sq[CQ][CDH];
  __shared__ float racc[CQ][2 + CDH];   // m, sum, o[64]
  for (int i = tid; i < CQ * CDH; i += 256) {
    int qq = i >> 6, d = i & 63;
    sq[qq][d] = bf2f(qbf[(((size_t)b * CNH + h) * CQ + qq) * CDH + d]);
  }
  if (tid < CQ) {
    racc[tid][0] = -1e30f; racc[tid][1] = 0.f;
#pragma unroll
    for (int d = 0; d < CDH; ++d) racc[tid][2 + d] = 0.f;
  }
  __syncthreads();

  float qreg[CDH];
#pragma unroll
  for (int d = 0; d < CDH; ++d) qreg[d] = sq[q][d];

  float m = -1e30f, ssum = 0.f, o[CDH];
#pragma unroll
  for (int d = 0; d < CDH; ++d) o[d] = 0.f;

  const int kmax = CL - CQ + q;
  const int k0 = stripe * (CL / 16), k1 = k0 + (CL / 16);
  for (int l = k0; l < k1; ++l) {
    if (l <= kmax) {
      const u32* kr = (const u32*)(kbf + (((size_t)b * CNH + h) * CL + l) * CDH);
      float s = 0.f;
#pragma unroll
      for (int dw = 0; dw < 32; ++dw) {
        u32 pk = kr[dw];
        s += qreg[2 * dw] * bf2f((u16)pk) + qreg[2 * dw + 1] * bf2f((u16)(pk >> 16));
      }
      s *= 0.125f;                       // 1/sqrt(64)
      float mn = fmaxf(m, s);
      float corr = __expf(m - mn);
      float p = __expf(s - mn);
      ssum = ssum * corr + p;
      const u32* vr = (const u32*)(vbf + (((size_t)b * CNH + h) * CL + l) * CDH);
#pragma unroll
      for (int dw = 0; dw < 32; ++dw) {
        u32 pv = vr[dw];
        o[2 * dw]     = o[2 * dw]     * corr + p * bf2f((u16)pv);
        o[2 * dw + 1] = o[2 * dw + 1] * corr + p * bf2f((u16)(pv >> 16));
      }
      m = mn;
    }
  }
  // serial merge of 16 stripes (distinct q per thread within a round)
  for (int si = 0; si < 16; ++si) {
    if (stripe == si) {
      float M0 = racc[q][0];
      float mn = fmaxf(M0, m);
      float c0 = __expf(M0 - mn), c1 = __expf(m - mn);
      racc[q][0] = mn;
      racc[q][1] = racc[q][1] * c0 + ssum * c1;
#pragma unroll
      for (int d = 0; d < CDH; ++d)
        racc[q][2 + d] = racc[q][2 + d] * c0 + o[d] * c1;
    }
    __syncthreads();
  }
  // out layout (B,Q,H,dh) == (B,Q,Din)
  for (int i = tid; i < CQ * CDH; i += 256) {
    int qq = i >> 6, d = i & 63;
    out[(((size_t)b * CQ + qq) * CNH + h) * CDH + d] = racc[qq][2 + d] / racc[qq][1];
  }
}

// NLM: shift 16-deep history, append pre = pre_raw + z, tiny per-neuron MLP.
__global__ void k_nlm(const float* __restrict__ pre_raw, const float* __restrict__ zin,
                      float* __restrict__ hist,
                      const float* __restrict__ w1, const float* __restrict__ b1,
                      const float* __restrict__ w2, const float* __restrict__ b2,
                      float* __restrict__ zout) {
  int idx = blockIdx.x * 256 + threadIdx.x;     // over B*Q*W
  if (idx >= CBQ * CW) return;
  int w = idx % CW;
  float pre = pre_raw[idx] + zin[idx];
  float* hrow = hist + (size_t)idx * CDD;
  float hv[CDD];
#pragma unroll
  for (int d = 0; d < CDD - 1; ++d) hv[d] = hrow[d + 1];
  hv[CDD - 1] = pre;
#pragma unroll
  for (int d = 0; d < CDD; ++d) hrow[d] = hv[d];
  float z = b2[w];
#pragma unroll
  for (int hh = 0; hh < 4; ++hh) {
    float a = b1[w * 4 + hh];
#pragma unroll
    for (int d = 0; d < CDD; ++d) a += hv[d] * w1[((size_t)w * CDD + d) * 4 + hh];
    z += gelu_f(a) * w2[w * 4 + hh];
  }
  zout[idx] = z;
}

// Synchronization features: per (b,q,s) exp-decay-weighted correlation.
__global__ void k_sync(const float* __restrict__ ring, int Tc, int lastSlot,
                       const int* __restrict__ il, const int* __restrict__ ir,
                       const float* __restrict__ decay, u16* __restrict__ out_bf) {
  int idx = blockIdx.x * 256 + threadIdx.x;     // over B*Q*S
  if (idx >= CBQ * CS) return;
  int s = idx % CS, bq = idx / CS;
  int a = il[s], r = ir[s];
  float dec = decay[s];
  float num = 0.f, wsum = 0.f;
  for (int t = 0; t < Tc; ++t) {
    int slot = (lastSlot + 8 - (Tc - 1) + t) & 7;
    const float* zt = ring + (size_t)slot * (CBQ * CW) + (size_t)bq * CW;
    float wgt = __expf(-dec * (float)(Tc - 1 - t));
    num += zt[a] * zt[r] * wgt;
    wsum += wgt;
  }
  out_bf[idx] = f2bf(num * rsqrtf(wsum));
}

// ---------------------------------------------------------------------------
// Host orchestration
// ---------------------------------------------------------------------------
extern "C" void kernel_launch(void* const* d_in, const int* in_sizes, int n_in,
                              void* d_out, int out_size, void* d_ws, size_t ws_size,
                              hipStream_t stream) {
  const int*   x        = (const int*)  d_in[0];
  const float* emb      = (const float*)d_in[2];
  const float* kv_w     = (const float*)d_in[3];
  const float* q_w      = (const float*)d_in[4];
  const float* o_w      = (const float*)d_in[5];
  const float* syn_w1   = (const float*)d_in[6];
  const float* syn_b1   = (const float*)d_in[7];
  const float* syn_w2   = (const float*)d_in[8];
  const float* syn_b2   = (const float*)d_in[9];
  const float* nlm_w1   = (const float*)d_in[10];
  const float* nlm_b1   = (const float*)d_in[11];
  const float* nlm_w2   = (const float*)d_in[12];
  const float* nlm_b2   = (const float*)d_in[13];
  const float* out_w    = (const float*)d_in[14];
  const float* out_b    = (const float*)d_in[15];
  const float* z_init   = (const float*)d_in[16];
  const float* hist0    = (const float*)d_in[17];
  const float* decay_a  = (const float*)d_in[18];
  const float* decay_o  = (const float*)d_in[19];
  const int*   idx_la   = (const int*)  d_in[20];
  const int*   idx_ra   = (const int*)  d_in[21];
  const int*   idx_lo   = (const int*)  d_in[22];
  const int*   idx_ro   = (const int*)  d_in[23];
  float* out = (float*)d_out;

  char* ws = (char*)d_ws;
  size_t off = 0;
  auto alloc = [&](size_t bytes) -> char* {
    char* p = ws + off;
    off += (bytes + 255) & ~(size_t)255;
    return p;
  };

  u16* kv_wT  = (u16*)alloc((size_t)CDIN * CDIN * 2);
  u16* q_wT   = (u16*)alloc((size_t)CDIN * CS   * 2);
  u16* o_wT   = (u16*)alloc((size_t)CDIN * CDIN * 2);
  u16* w1T    = (u16*)alloc((size_t)CHS  * CCAT * 2);
  u16* w2T    = (u16*)alloc((size_t)CW   * CHS  * 2);
  u16* outT   = (u16*)alloc((size_t)CV   * CS   * 2);
  u16* aemb   = (u16*)alloc((size_t)CB * CL * CDIN * 2);
  float* kvf  = (float*)alloc((size_t)CB * CL * CDIN * 4);
  u16* kbf    = (u16*)alloc((size_t)CB * CNH * CL * CDH * 2);
  u16* vbf    = (u16*)alloc((size_t)CB * CNH * CL * CDH * 2);
  float* hist = (float*)alloc((size_t)CBQ * CW * CDD * 4);
  float* ring = (float*)alloc((size_t)8 * CBQ * CW * 4);
  u16* syncA  = (u16*)alloc((size_t)CBQ * CS * 2);
  u16* syncO  = (u16*)alloc((size_t)CBQ * CS * 2);
  float* qf   = (float*)alloc((size_t)CBQ * CDIN * 4);
  u16* qbf    = (u16*)alloc((size_t)CB * CNH * CQ * CDH * 2);
  float* attnf  = (float*)alloc((size_t)CBQ * CDIN * 4);
  u16* attnbf   = (u16*)alloc((size_t)CBQ * CDIN * 2);
  float* attnpf = (float*)alloc((size_t)CBQ * CDIN * 4);
  u16* catbf    = (u16*)alloc((size_t)CBQ * CCAT * 2);
  float* hf     = (float*)alloc((size_t)CBQ * CHS * 4);
  u16* hbf      = (u16*)alloc((size_t)CBQ * CHS * 2);
  float* pref   = (float*)alloc((size_t)CBQ * CW * 4);

  auto blocks = [](size_t n) { return (unsigned)((n + 255) / 256); };
  auto gemm = [&](const u16* A, const u16* BT, const float* bias, float* C,
                  int M, int N, int K, int ab) {
    k_gemm_bf16<<<dim3((unsigned)((M / 128) * (N / 64))), dim3(128), 0, stream>>>(
        A, BT, bias, C, M, N, K, ab);
  };

  // ---- one-time setup ----
  k_transpose_bf16<<<blocks((size_t)CDIN * CDIN), 256, 0, stream>>>(kv_w, kv_wT, CDIN, CDIN);
  k_transpose_bf16<<<blocks((size_t)CS * CDIN),   256, 0, stream>>>(q_w,  q_wT,  CS,   CDIN);
  k_transpose_bf16<<<blocks((size_t)CDIN * CDIN), 256, 0, stream>>>(o_w,  o_wT,  CDIN, CDIN);
  k_transpose_bf16<<<blocks((size_t)CCAT * CHS),  256, 0, stream>>>(syn_w1, w1T, CCAT, CHS);
  k_transpose_bf16<<<blocks((size_t)CHS * CW),    256, 0, stream>>>(syn_w2, w2T, CHS,  CW);
  k_transpose_bf16<<<blocks((size_t)CS * CV),     256, 0, stream>>>(out_w, outT, CS,   CV);

  k_gather_embed<<<blocks((size_t)CB * CL * CDIN), 256, 0, stream>>>(x, emb, aemb);
  gemm(aemb, kv_wT, nullptr, kvf, CB * CL, CDIN, CDIN, 0);
  k_rope_kv<<<blocks((size_t)CB * CL * CNH * 32), 256, 0, stream>>>(kvf, kbf, vbf);

  k_init_hist<<<blocks((size_t)CBQ * CW * CDD), 256, 0, stream>>>(hist0, hist);
  k_init_post<<<blocks((size_t)CBQ * CW), 256, 0, stream>>>(z_init, ring);
  k_sync<<<blocks((size_t)CBQ * CS), 256, 0, stream>>>(ring, 1, 0, idx_la, idx_ra,
                                                       decay_a, syncA);

  // ---- thought steps ----
  for (int i = 0; i < CT; ++i) {
    float* zcur = ring + (size_t)(i & 7) * (CBQ * CW);
    float* znew = ring + (size_t)((i + 1) & 7) * (CBQ * CW);

    gemm(syncA, q_wT, nullptr, qf, CBQ, CDIN, CS, 0);
    k_rope_q<<<blocks((size_t)CB * CQ * CNH * 32), 256, 0, stream>>>(qf, qbf);
    k_attention<<<dim3(CB * CNH), 256, 0, stream>>>(qbf, kbf, vbf, attnf);
    k_cast_bf16<<<blocks((size_t)CBQ * CDIN), 256, 0, stream>>>(attnf, attnbf, CBQ * CDIN);
    gemm(attnbf, o_wT, nullptr, attnpf, CBQ, CDIN, CDIN, 0);

    k_concat_cast<<<blocks((size_t)CBQ * CCAT), 256, 0, stream>>>(attnpf, zcur, catbf);
    gemm(catbf, w1T, syn_b1, hf, CBQ, CHS, CCAT, 1);
    k_gelu_cast<<<blocks((size_t)CBQ * CHS), 256, 0, stream>>>(hf, hbf, CBQ * CHS);
    gemm(hbf, w2T, syn_b2, pref, CBQ, CW, CHS, 1);

    k_nlm<<<blocks((size_t)CBQ * CW), 256, 0, stream>>>(pref, zcur, hist, nlm_w1,
                                                        nlm_b1, nlm_w2, nlm_b2, znew);

    int Tc = (i + 2 < 8) ? (i + 2) : 8;
    int last = (i + 1) & 7;
    k_sync<<<blocks((size_t)CBQ * CS), 256, 0, stream>>>(ring, Tc, last, idx_la, idx_ra,
                                                         decay_a, syncA);
    k_sync<<<blocks((size_t)CBQ * CS), 256, 0, stream>>>(ring, Tc, last, idx_lo, idx_ro,
                                                         decay_o, syncO);
    gemm(syncO, outT, out_b, out + (size_t)i * CBQ * CV, CBQ, CV, CS, 1);
  }
  (void)in_sizes; (void)n_in; (void)out_size; (void)ws_size;
}
